// AttentionBlock_28028956574020
// MI455X (gfx1250) — matrix operand order repaired and checked by reading, not compiled, not run on hardware
//
#include <hip/hip_runtime.h>

typedef unsigned int u32;
typedef unsigned short u16;
typedef __bf16 v16bf __attribute__((ext_vector_type(16)));
typedef float v8f __attribute__((ext_vector_type(8)));

#define BB 8
#define LL 2048
#define DD 1024
#define GG 4
#define DH 256
#define MM (BB * LL)   // 16384 flat rows
#define NEGF (-1.0e30f)

union Frag {
    v16bf v;
    uint4 q[2];
    u32   w[8];
    u16   u[16];
};

__device__ __forceinline__ v8f wmma_bf(const Frag& a, const Frag& b, v8f c) {
    return __builtin_amdgcn_wmma_f32_16x16x32_bf16(false, a.v, false, b.v,
                                                   (short)0, c, false, false);
}

__device__ __forceinline__ u16 f2bf(float x) {
    u32 u = __float_as_uint(x);
    return (u16)((u + 0x7FFFu + ((u >> 16) & 1u)) >> 16);
}

// ---------------------------------------------------------------- convert f32 -> bf16
__global__ __launch_bounds__(256) void cvt_bf16_kernel(const float* __restrict__ in,
                                                       u16* __restrict__ out, int n) {
    int i = (blockIdx.x * 256 + threadIdx.x) * 4;
    if (i + 3 < n) {
        float4 v = *(const float4*)(in + i);
        u32 lo = (u32)f2bf(v.x) | ((u32)f2bf(v.y) << 16);
        u32 hi = (u32)f2bf(v.z) | ((u32)f2bf(v.w) << 16);
        uint2 p; p.x = lo; p.y = hi;
        *(uint2*)(out + i) = p;
    }
}

// ---------------------------------------------------------------- generic K=1024 GEMM
// C[m,n] = A[m,:] dot W[n,:]  (both bf16, row-major, K contiguous)
// Wave tile 32x64 (2 M-tiles x 4 N-tiles), double-buffered register fragments.
// MODE 0: out0 <- bf16(C)                           (q_in projection)
// MODE 1: C += bias; split n: [0,1024)->qb, [1024,2048)->kb, [2048,3072)->vtb (transposed)
// MODE 2: A gets ReLU on load; hout <- f32(C)       (offset predictor hidden)
template <int MODE>
__global__ __launch_bounds__(256) void gemm_bf16_k1024(
    const u16* __restrict__ A, const u16* __restrict__ W, const float* __restrict__ bias,
    u16* __restrict__ out0, u16* __restrict__ qb, u16* __restrict__ kb,
    u16* __restrict__ vtb, float* __restrict__ hout)
{
    const int tid  = threadIdx.x;
    const int wave = tid >> 5, lane = tid & 31, ln = lane & 15, hh = lane >> 4;
    const int m0 = blockIdx.x * 256 + wave * 32;
    const int n0 = blockIdx.y * 64;

    const u16* arow0 = A + (size_t)(m0 + ln) * 1024;
    const u16* arow1 = A + (size_t)(m0 + 16 + ln) * 1024;
    const u16* wbase = W + (size_t)(n0 + ln) * 1024;

    v8f acc[2][4];
    const v8f vzero = {0.f, 0.f, 0.f, 0.f, 0.f, 0.f, 0.f, 0.f};
#pragma unroll
    for (int mi = 0; mi < 2; ++mi)
#pragma unroll
        for (int nt = 0; nt < 4; ++nt) acc[mi][nt] = vzero;

    Frag fa[2][2], fb[2][4];

    auto load_step = [&](int s, int p) {
        const u16* ap0 = arow0 + s * 32;
        const u16* ap1 = arow1 + s * 32;
        __builtin_prefetch(ap0 + 512, 0, 1);               // global_prefetch_b8
        fa[p][0].q[0] = *(const uint4*)(ap0 + hh * 8);      // A frag: K = h*8..+7
        fa[p][0].q[1] = *(const uint4*)(ap0 + 16 + hh * 8); //         K = 16+h*8..+7
        fa[p][1].q[0] = *(const uint4*)(ap1 + hh * 8);
        fa[p][1].q[1] = *(const uint4*)(ap1 + 16 + hh * 8);
        if constexpr (MODE == 2) {                          // bf16 ReLU, bitwise
#pragma unroll
            for (int mi = 0; mi < 2; ++mi)
#pragma unroll
                for (int e = 0; e < 8; ++e) {
                    u32 v = fa[p][mi].w[e];
                    u32 sgn = (v >> 15) & 0x00010001u;
                    fa[p][mi].w[e] = v & ~(sgn * 0x0000FFFFu);
                }
        }
#pragma unroll
        for (int nt = 0; nt < 4; ++nt) {
            const u16* wp = wbase + (size_t)nt * 16 * 1024 + s * 32;
            fb[p][nt].q[0] = *(const uint4*)(wp + hh * 8);      // B frag: K = h*8..+7, 16+h*8..+7
            fb[p][nt].q[1] = *(const uint4*)(wp + 16 + hh * 8);
        }
    };

    load_step(0, 0);
#pragma unroll 2
    for (int s = 0; s < 32; ++s) {
        const int p = s & 1;
        load_step((s + 1) & 31, p ^ 1);   // next step's loads fly behind WMMAs
#pragma unroll
        for (int mi = 0; mi < 2; ++mi)
#pragma unroll
            for (int nt = 0; nt < 4; ++nt)
                acc[mi][nt] = wmma_bf(fa[p][mi], fb[p][nt], acc[mi][nt]);
    }

#pragma unroll
    for (int mi = 0; mi < 2; ++mi)
#pragma unroll
        for (int nt = 0; nt < 4; ++nt)
#pragma unroll
            for (int r = 0; r < 8; ++r) {
                float c = acc[mi][nt][r];
                int m = m0 + mi * 16 + hh * 8 + r;
                int n = n0 + nt * 16 + ln;
                if constexpr (MODE == 0) {
                    out0[(size_t)m * 1024 + n] = f2bf(c);
                } else if constexpr (MODE == 1) {
                    c += bias[n];
                    int sec = n >> 10, d = n & 1023;
                    if (sec == 0) {
                        qb[(size_t)m * 1024 + d] = f2bf(c);
                    } else if (sec == 1) {
                        kb[(size_t)m * 1024 + d] = f2bf(c);
                    } else {
                        int g = d >> 8, nn = d & 255;
                        int b = m >> 11, ml = m & 2047;
                        vtb[((size_t)(b * GG + g) * DH + nn) * LL + ml] = f2bf(c);
                    }
                } else {
                    hout[(size_t)m * 256 + n] = c;
                }
            }
}

// ---------------------------------------------------------------- dx = sigmoid(h.W2 + b2)*L
__global__ __launch_bounds__(256) void dx_kernel(const float* __restrict__ h,
                                                 const float* __restrict__ W2,
                                                 const float* __restrict__ b2,
                                                 float* __restrict__ dx) {
    int m = blockIdx.x * 256 + threadIdx.x;
    if (m >= MM) return;
    const float* hr = h + (size_t)m * 256;
    float a = 0.f;
#pragma unroll 8
    for (int k = 0; k < 256; ++k) a += hr[k] * W2[k];
    a += b2[0];
    dx[m] = (float)LL / (1.f + __expf(-a));
}

// ---------------------------------------------------------------- fused deformable flash attention
// grid.x = B*G, grid.y = L/128. 8 waves, each owns 16 query rows, full dh=256 acc.
// Register budget kept < 256 VGPRs: Q fragments are re-read from L0 each key tile,
// K/V^T staging is direct global->LDS, next tile is warmed via global_prefetch_b8.
__global__ __launch_bounds__(256) void attn_kernel(const u16* __restrict__ qb,
                                                   const u16* __restrict__ kb,
                                                   const u16* __restrict__ vtb,
                                                   const float* __restrict__ dx,
                                                   float* __restrict__ out)
{
    __shared__ __align__(16) u16 kt[32 * 264];        // K tile: 32 keys x 256 dh (+8 pad)
    __shared__ __align__(16) u16 vt[256 * 40];        // V^T tile: 256 dh x 32 keys (+8 pad)
    __shared__ __align__(16) u16 ps[8 * 16 * 40];     // P tiles: per-wave 16 x 32 (+8 pad)
    __shared__ float sred[8];

    const int tid  = threadIdx.x;
    const int wave = tid >> 5, lane = tid & 31, ln = lane & 15, hh = lane >> 4;
    const int bg = blockIdx.x, b = bg >> 2, g = bg & 3;
    const int ibase = blockIdx.y * 128;
    const int m0 = ibase + wave * 16;

    const float wg  = (g == 0) ? 32.f : (g == 1) ? 64.f : (g == 2) ? 128.f : 256.f;
    const float sc2 = 0.04419417382415922f;           // 2 / sqrt(2048): doubled scores

    const u16* qrow = qb + (size_t)(b * LL + m0 + ln) * DD + g * DH;

    // ---- per-row window size, softmax state, output accumulators
    float dxr[8], rm[8], rs[8];
#pragma unroll
    for (int r = 0; r < 8; ++r) {
        dxr[r] = dx[b * LL + m0 + hh * 8 + r] + wg;
        rm[r] = NEGF;
        rs[r] = 0.f;
    }
    v8f acc[16];
    const v8f vzero = {0.f, 0.f, 0.f, 0.f, 0.f, 0.f, 0.f, 0.f};
#pragma unroll
    for (int nt = 0; nt < 16; ++nt) acc[nt] = vzero;

    // ---- block-uniform end of key range (early exit)
    float mx = dxr[0];
#pragma unroll
    for (int r = 1; r < 8; ++r) mx = fmaxf(mx, dxr[r]);
    for (int off = 16; off >= 1; off >>= 1) mx = fmaxf(mx, __shfl_xor(mx, off, 32));
    if (lane == 0) sred[wave] = mx;
    __syncthreads();
    float bm = sred[0];
#pragma unroll
    for (int w = 1; w < 8; ++w) bm = fmaxf(bm, sred[w]);
    int jend = ibase + 127 + (int)bm + 2;
    if (jend > LL) jend = LL;

    for (int j0 = ibase; j0 < jend; j0 += 32) {
        __syncthreads();
        // ---- stage K tile (row-major) and V^T tile into LDS cooperatively
#pragma unroll
        for (int k = 0; k < 4; ++k) {
            int c = tid + k * 256;
            int row = c >> 5, colk = (c & 31) << 3;
            *(uint4*)(&kt[row * 264 + colk]) =
                *(const uint4*)(kb + (size_t)(b * LL + j0 + row) * DD + g * DH + colk);
            int n = c >> 2, colv = (c & 3) << 3;
            *(uint4*)(&vt[n * 40 + colv]) =
                *(const uint4*)(vtb + (size_t)(bg * DH + n) * LL + j0 + colv);
        }
        __syncthreads();

        // ---- warm next tile's cachelines during compute (no registers held)
        int jp = (j0 + 32 < jend) ? (j0 + 32) : j0;
        if (tid < 128) {   // K tile: 32 rows x 512B = 128 lines of 128B
            int row = tid >> 2, col = (tid & 3) << 6;
            __builtin_prefetch(kb + (size_t)(b * LL + jp + row) * DD + g * DH + col, 0, 1);
        } else {           // V^T tile: 256 rows x 64B -> prefetch every other row head
            int n = (tid - 128) << 1;
            __builtin_prefetch(vtb + (size_t)(bg * DH + n) * LL + jp, 0, 1);
        }

        // ---- S = Q K^T over 32 keys (two 16-wide n-tiles); Q frags re-read from L0
        v8f S0 = vzero, S1 = vzero;
#pragma unroll
        for (int s = 0; s < 8; ++s) {
            Frag fq, fk0, fk1;
            fq.q[0] = *(const uint4*)(qrow + s * 32 + hh * 8);
            fq.q[1] = *(const uint4*)(qrow + s * 32 + 16 + hh * 8);
            const u16* k0 = &kt[ln * 264 + s * 32];
            const u16* k1 = &kt[(16 + ln) * 264 + s * 32];
            fk0.q[0] = *(const uint4*)(k0 + hh * 8);
            fk0.q[1] = *(const uint4*)(k0 + 16 + hh * 8);
            fk1.q[0] = *(const uint4*)(k1 + hh * 8);
            fk1.q[1] = *(const uint4*)(k1 + 16 + hh * 8);
            S0 = wmma_bf(fq, fk0, S0);
            S1 = wmma_bf(fq, fk1, S1);
        }

        // ---- mask + online softmax (rows live in 16-lane halves)
#pragma unroll
        for (int r = 0; r < 8; ++r) {
            int i = m0 + hh * 8 + r;
            int jg0 = j0 + ln, jg1 = j0 + 16 + ln;
            bool ok0 = (jg0 >= i) && ((float)(jg0 - i) <= dxr[r]);
            bool ok1 = (jg1 >= i) && ((float)(jg1 - i) <= dxr[r]);
            float s0 = ok0 ? S0[r] * sc2 : NEGF;
            float s1 = ok1 ? S1[r] * sc2 : NEGF;
            float tm = fmaxf(s0, s1);
            for (int off = 8; off >= 1; off >>= 1) tm = fmaxf(tm, __shfl_xor(tm, off, 16));
            float nm = fmaxf(rm[r], tm);
            float fac = __expf(rm[r] - nm);
            float p0 = ok0 ? __expf(s0 - nm) : 0.f;
            float p1 = ok1 ? __expf(s1 - nm) : 0.f;
            float ts = p0 + p1;
            for (int off = 8; off >= 1; off >>= 1) ts += __shfl_xor(ts, off, 16);
            rm[r] = nm;
            rs[r] = rs[r] * fac + ts;
#pragma unroll
            for (int nt = 0; nt < 16; ++nt) acc[nt][r] *= fac;
            ps[wave * 640 + (hh * 8 + r) * 40 + ln] = f2bf(p0);
            ps[wave * 640 + (hh * 8 + r) * 40 + 16 + ln] = f2bf(p1);
        }

        // intra-wave LDS store->load handoff for P
        asm volatile("s_wait_dscnt 0" ::: "memory");

        Frag fp;  // P as A-operand
        const u16* pr = &ps[wave * 640 + ln * 40];
        fp.q[0] = *(const uint4*)(pr + hh * 8);
        fp.q[1] = *(const uint4*)(pr + 16 + hh * 8);

        // ---- acc += P V, over all 16 dh n-tiles
#pragma unroll
        for (int nt = 0; nt < 16; ++nt) {
            Frag fv;
            const u16* vr = &vt[(nt * 16 + ln) * 40];
            fv.q[0] = *(const uint4*)(vr + hh * 8);
            fv.q[1] = *(const uint4*)(vr + 16 + hh * 8);
            acc[nt] = wmma_bf(fp, fv, acc[nt]);
        }
    }

    // ---- normalize and write output (f32)
#pragma unroll
    for (int r = 0; r < 8; ++r) {
        int i = m0 + hh * 8 + r;
        float inv = 1.0f / rs[r];
        float* orow = out + (size_t)(b * LL + i) * DD + g * DH;
#pragma unroll
        for (int nt = 0; nt < 16; ++nt) orow[nt * 16 + ln] = acc[nt][r] * inv;
    }
}

// ---------------------------------------------------------------- host launcher
extern "C" void kernel_launch(void* const* d_in, const int* in_sizes, int n_in,
                              void* d_out, int out_size, void* d_ws, size_t ws_size,
                              hipStream_t stream) {
    const float* x    = (const float*)d_in[0];
    const float* Wq   = (const float*)d_in[1];
    // d_in[2]=Wk, d_in[3]=Wv : dead compute in reference, skipped
    const float* Wqkv = (const float*)d_in[4];
    const float* bqkv = (const float*)d_in[5];
    const float* Wop  = (const float*)d_in[6];
    const float* W2   = (const float*)d_in[7];
    const float* b2   = (const float*)d_in[8];
    float* out = (float*)d_out;

    char* ws = (char*)d_ws;
    size_t off = 0;
    auto carve = [&](size_t bytes) {
        char* p = ws + off;
        off += (bytes + 255) & ~(size_t)255;
        return p;
    };
    u16* x_bf    = (u16*)carve((size_t)MM * DD * 2);
    u16* qin_bf  = (u16*)carve((size_t)MM * DD * 2);
    u16* q_bf    = (u16*)carve((size_t)MM * DD * 2);
    u16* k_bf    = (u16*)carve((size_t)MM * DD * 2);
    u16* vT_bf   = (u16*)carve((size_t)MM * DD * 2);
    u16* Wq_bf   = (u16*)carve((size_t)DD * DD * 2);
    u16* Wqkv_bf = (u16*)carve((size_t)3 * DD * DD * 2);
    u16* Wop_bf  = (u16*)carve((size_t)256 * DD * 2);
    float* h_f   = (float*)carve((size_t)MM * 256 * 4);
    float* dx_f  = (float*)carve((size_t)MM * 4);
    (void)ws_size; (void)n_in; (void)in_sizes; (void)out_size;

    // 1) convert inputs/weights to bf16
    cvt_bf16_kernel<<<(MM * DD) / 1024, 256, 0, stream>>>(x, x_bf, MM * DD);
    cvt_bf16_kernel<<<(DD * DD) / 1024, 256, 0, stream>>>(Wq, Wq_bf, DD * DD);
    cvt_bf16_kernel<<<(3 * DD * DD) / 1024, 256, 0, stream>>>(Wqkv, Wqkv_bf, 3 * DD * DD);
    cvt_bf16_kernel<<<(256 * DD) / 1024, 256, 0, stream>>>(Wop, Wop_bf, 256 * DD);

    // 2) q_in = x @ Wq^T
    gemm_bf16_k1024<0><<<dim3(MM / 256, DD / 64), 256, 0, stream>>>(
        x_bf, Wq_bf, nullptr, qin_bf, nullptr, nullptr, nullptr, nullptr);

    // 3) qkv = q_in @ Wqkv^T + b  -> q, k, v^T
    gemm_bf16_k1024<1><<<dim3(MM / 256, (3 * DD) / 64), 256, 0, stream>>>(
        qin_bf, Wqkv_bf, bqkv, nullptr, q_bf, k_bf, vT_bf, nullptr);

    // 4) h = relu(q_in) @ Wop^T ; dx = sigmoid(h @ W2^T + b2) * L
    gemm_bf16_k1024<2><<<dim3(MM / 256, 256 / 64), 256, 0, stream>>>(
        qin_bf, Wop_bf, nullptr, nullptr, nullptr, nullptr, nullptr, h_f);
    dx_kernel<<<MM / 256, 256, 0, stream>>>(h_f, W2, b2, dx_f);

    // 5) fused windowed flash attention
    attn_kernel<<<dim3(BB * GG, LL / 128), 256, 0, stream>>>(q_bf, k_bf, vT_bf, dx_f, out);
}
